// MIL_Cluster_FC_15857019257485
// MI455X (gfx1250) — compile-verified, matrix operand order checked by reading, not run on hardware
//
#include <hip/hip_runtime.h>
#include <hip/hip_bf16.h>
#include <math.h>

typedef __bf16 bf16_t;
typedef __attribute__((ext_vector_type(16))) __bf16 v16bf;
typedef __attribute__((ext_vector_type(8)))  __bf16 v8bf;
typedef __attribute__((ext_vector_type(8)))  float  v8f;

#define N_PTS   10000
#define D_IN    1024
#define D_H     512
#define D_ATT   256
#define C_CLUST 10
#define N_CLS   4
#define TILES_M32 ((N_PTS + 31) / 32)   // 313 row tiles of 32

// ---------------- workspace layout (bytes) ----------------
constexpr size_t SZ_XB   = (size_t)N_PTS * D_IN * 2;            // x in bf16
constexpr size_t SZ_W1T  = (size_t)C_CLUST * D_H * D_IN * 2;    // W1^T bf16 [c][col][k]
constexpr size_t SZ_W2T  = (size_t)C_CLUST * D_H * D_H * 2;     // W2^T bf16 [c][col][k]
constexpr size_t SZ_IDX  = (size_t)C_CLUST * N_PTS * 4;         // per-cluster ordered indices
constexpr size_t SZ_H1   = (size_t)N_PTS * D_H * 2;             // compacted h1 bf16
constexpr size_t SZ_PART = (size_t)C_CLUST * TILES_M32 * D_H * 4; // per-tile col sums

constexpr size_t OFF_XB   = 0;
constexpr size_t OFF_W1T  = OFF_XB + SZ_XB;
constexpr size_t OFF_W2T  = OFF_W1T + SZ_W1T;
constexpr size_t OFF_IDX  = OFF_W2T + SZ_W2T;
constexpr size_t OFF_CNT  = OFF_IDX + SZ_IDX;
constexpr size_t OFF_BASE = OFF_CNT + 256;
constexpr size_t OFF_H1   = OFF_BASE + 256;
constexpr size_t OFF_PART = OFF_H1 + SZ_H1;

__device__ __forceinline__ float relu_f(float x) { return x > 0.f ? x : 0.f; }

// Build a 16x32 bf16 WMMA fragment for this lane: elements 0..7 = K 8h..8h+7,
// elements 8..15 = K 16+8h..16+8h+7 (ISA 7.12.2 16-bit A layout; B symmetric).
__device__ __forceinline__ v16bf ldfrag(const bf16_t* __restrict__ p, int hh) {
    v8bf lo = *(const v8bf*)(p + 8 * hh);
    v8bf hi = *(const v8bf*)(p + 16 + 8 * hh);
    return __builtin_shufflevector(lo, hi, 0,1,2,3,4,5,6,7,8,9,10,11,12,13,14,15);
}

// ---------------- conversion / transpose kernels ----------------
__global__ void k_cvt_x(const float* __restrict__ x, bf16_t* __restrict__ xb, int n) {
    int i = blockIdx.x * blockDim.x + threadIdx.x;
    if (i < n) xb[i] = (bf16_t)x[i];
}

// W1: [C][D_IN][D_H] f32 -> w1t: [C][D_H][D_IN] bf16 (K contiguous per output col)
__global__ void k_cvt_w1(const float* __restrict__ w, bf16_t* __restrict__ wt) {
    int i = blockIdx.x * blockDim.x + threadIdx.x;
    int total = C_CLUST * D_H * D_IN;
    if (i >= total) return;
    int k = i % D_IN;
    int col = (i / D_IN) % D_H;
    int c = i / (D_H * D_IN);
    wt[i] = (bf16_t)w[((size_t)c * D_IN + k) * D_H + col];
}

// W2: [C][D_H][D_H] f32 -> w2t: [C][D_H][D_H] bf16, transposed
__global__ void k_cvt_w2(const float* __restrict__ w, bf16_t* __restrict__ wt) {
    int i = blockIdx.x * blockDim.x + threadIdx.x;
    int total = C_CLUST * D_H * D_H;
    if (i >= total) return;
    int k = i % D_H;
    int col = (i / D_H) % D_H;
    int c = i / (D_H * D_H);
    wt[i] = (bf16_t)w[((size_t)c * D_H + k) * D_H + col];
}

// ---------------- deterministic bucketing (stable order) ----------------
__global__ void k_bucket(const int* __restrict__ cid, int* __restrict__ idx,
                         int* __restrict__ cnt) {
    int c = blockIdx.x;
    __shared__ int sf[256];
    __shared__ int sbase;
    if (threadIdx.x == 0) sbase = 0;
    __syncthreads();
    for (int s0 = 0; s0 < N_PTS; s0 += 256) {
        int i = s0 + (int)threadIdx.x;
        int f = (i < N_PTS && cid[i] == c) ? 1 : 0;
        sf[threadIdx.x] = f;
        __syncthreads();
        for (int off = 1; off < 256; off <<= 1) {
            int v = (threadIdx.x >= (unsigned)off) ? sf[threadIdx.x - off] : 0;
            __syncthreads();
            sf[threadIdx.x] += v;
            __syncthreads();
        }
        if (f) idx[c * N_PTS + sbase + sf[threadIdx.x] - 1] = i;
        __syncthreads();
        if (threadIdx.x == 0) sbase += sf[255];
        __syncthreads();
    }
    if (threadIdx.x == 0) cnt[c] = sbase;
}

__global__ void k_base(const int* __restrict__ cnt, int* __restrict__ base) {
    if (blockIdx.x == 0 && threadIdx.x == 0) {
        int s = 0;
        for (int c = 0; c < C_CLUST; ++c) { base[c] = s; s += cnt[c]; }
    }
}

// ---------------- layer 1: h1 = ReLU(Xg @ W1[c] + b1[c]) ----------------
// grid: (2, TILES_M32, C_CLUST), block = 128 (4 waves)
// Each wave: 32 rows x 64 cols = 8 accumulators; per K=32 step:
// 2 A-frags + 4 B-frags (12 b128 loads) feed 8 WMMAs (1.5 loads/WMMA).
__global__ void k_phi1(const bf16_t* __restrict__ xb, const bf16_t* __restrict__ w1t,
                       const float* __restrict__ b1, const int* __restrict__ idx,
                       const int* __restrict__ cnt, const int* __restrict__ base,
                       bf16_t* __restrict__ h1) {
    int c = blockIdx.z;
    int tm = blockIdx.y;
    int n_c = cnt[c];
    if (tm * 32 >= n_c) return;
    int wave = threadIdx.x >> 5;
    int lane = threadIdx.x & 31;
    int m = lane & 15;
    int hh = lane >> 4;
    int colbase = blockIdx.x * 256 + wave * 64;

    // gather two A row groups (clamped; padding rows never stored)
    int r0 = tm * 32 + m;
    int r1 = r0 + 16;
    int g0 = idx[c * N_PTS + (r0 < n_c ? r0 : n_c - 1)];
    int g1 = idx[c * N_PTS + (r1 < n_c ? r1 : n_c - 1)];
    const bf16_t* arow0 = xb + (size_t)g0 * D_IN;
    const bf16_t* arow1 = xb + (size_t)g1 * D_IN;
    const bf16_t* brow0 = w1t + ((size_t)c * D_H + colbase + 0 * 16 + m) * D_IN;
    const bf16_t* brow1 = w1t + ((size_t)c * D_H + colbase + 1 * 16 + m) * D_IN;
    const bf16_t* brow2 = w1t + ((size_t)c * D_H + colbase + 2 * 16 + m) * D_IN;
    const bf16_t* brow3 = w1t + ((size_t)c * D_H + colbase + 3 * 16 + m) * D_IN;

    v8f acc[2][4];
#pragma unroll
    for (int a = 0; a < 2; ++a)
#pragma unroll
        for (int t = 0; t < 4; ++t)
            acc[a][t] = (v8f){0.f, 0.f, 0.f, 0.f, 0.f, 0.f, 0.f, 0.f};

    for (int k0 = 0; k0 < D_IN; k0 += 32) {
        v16bf a0 = ldfrag(arow0 + k0, hh);
        v16bf a1 = ldfrag(arow1 + k0, hh);
        v16bf bv0 = ldfrag(brow0 + k0, hh);
        v16bf bv1 = ldfrag(brow1 + k0, hh);
        v16bf bv2 = ldfrag(brow2 + k0, hh);
        v16bf bv3 = ldfrag(brow3 + k0, hh);
        acc[0][0] = __builtin_amdgcn_wmma_f32_16x16x32_bf16(false, a0, false, bv0, (short)0, acc[0][0], false, false);
        acc[1][0] = __builtin_amdgcn_wmma_f32_16x16x32_bf16(false, a1, false, bv0, (short)0, acc[1][0], false, false);
        acc[0][1] = __builtin_amdgcn_wmma_f32_16x16x32_bf16(false, a0, false, bv1, (short)0, acc[0][1], false, false);
        acc[1][1] = __builtin_amdgcn_wmma_f32_16x16x32_bf16(false, a1, false, bv1, (short)0, acc[1][1], false, false);
        acc[0][2] = __builtin_amdgcn_wmma_f32_16x16x32_bf16(false, a0, false, bv2, (short)0, acc[0][2], false, false);
        acc[1][2] = __builtin_amdgcn_wmma_f32_16x16x32_bf16(false, a1, false, bv2, (short)0, acc[1][2], false, false);
        acc[0][3] = __builtin_amdgcn_wmma_f32_16x16x32_bf16(false, a0, false, bv3, (short)0, acc[0][3], false, false);
        acc[1][3] = __builtin_amdgcn_wmma_f32_16x16x32_bf16(false, a1, false, bv3, (short)0, acc[1][3], false, false);
    }

    int bs = base[c];
#pragma unroll
    for (int t = 0; t < 4; ++t) {
        int col = colbase + t * 16 + m;
        float bias = b1[c * D_H + col];
#pragma unroll
        for (int a = 0; a < 2; ++a) {
#pragma unroll
            for (int v = 0; v < 8; ++v) {
                int row = tm * 32 + a * 16 + v + 8 * hh;   // D[m = v+8h] within subtile
                if (row < n_c) {
                    float val = relu_f(acc[a][t][v] + bias);
                    h1[(size_t)(bs + row) * D_H + col] = (bf16_t)val;
                }
            }
        }
    }
}

// ---------------- layer 2 + masked column sums ----------------
__global__ void k_phi2(const bf16_t* __restrict__ h1, const bf16_t* __restrict__ w2t,
                       const float* __restrict__ b2, const int* __restrict__ cnt,
                       const int* __restrict__ base, float* __restrict__ part) {
    int c = blockIdx.z;
    int tm = blockIdx.y;
    int n_c = cnt[c];
    if (tm * 32 >= n_c) return;
    int wave = threadIdx.x >> 5;
    int lane = threadIdx.x & 31;
    int m = lane & 15;
    int hh = lane >> 4;
    int colbase = blockIdx.x * 256 + wave * 64;

    int s0 = base[c] + tm * 32 + m;
    int s1 = s0 + 16;
    if (s0 > N_PTS - 1) s0 = N_PTS - 1;          // clamp; padding rows masked below
    if (s1 > N_PTS - 1) s1 = N_PTS - 1;
    const bf16_t* arow0 = h1 + (size_t)s0 * D_H;
    const bf16_t* arow1 = h1 + (size_t)s1 * D_H;
    const bf16_t* brow0 = w2t + ((size_t)c * D_H + colbase + 0 * 16 + m) * D_H;
    const bf16_t* brow1 = w2t + ((size_t)c * D_H + colbase + 1 * 16 + m) * D_H;
    const bf16_t* brow2 = w2t + ((size_t)c * D_H + colbase + 2 * 16 + m) * D_H;
    const bf16_t* brow3 = w2t + ((size_t)c * D_H + colbase + 3 * 16 + m) * D_H;

    v8f acc[2][4];
#pragma unroll
    for (int a = 0; a < 2; ++a)
#pragma unroll
        for (int t = 0; t < 4; ++t)
            acc[a][t] = (v8f){0.f, 0.f, 0.f, 0.f, 0.f, 0.f, 0.f, 0.f};

    for (int k0 = 0; k0 < D_H; k0 += 32) {
        v16bf a0 = ldfrag(arow0 + k0, hh);
        v16bf a1 = ldfrag(arow1 + k0, hh);
        v16bf bv0 = ldfrag(brow0 + k0, hh);
        v16bf bv1 = ldfrag(brow1 + k0, hh);
        v16bf bv2 = ldfrag(brow2 + k0, hh);
        v16bf bv3 = ldfrag(brow3 + k0, hh);
        acc[0][0] = __builtin_amdgcn_wmma_f32_16x16x32_bf16(false, a0, false, bv0, (short)0, acc[0][0], false, false);
        acc[1][0] = __builtin_amdgcn_wmma_f32_16x16x32_bf16(false, a1, false, bv0, (short)0, acc[1][0], false, false);
        acc[0][1] = __builtin_amdgcn_wmma_f32_16x16x32_bf16(false, a0, false, bv1, (short)0, acc[0][1], false, false);
        acc[1][1] = __builtin_amdgcn_wmma_f32_16x16x32_bf16(false, a1, false, bv1, (short)0, acc[1][1], false, false);
        acc[0][2] = __builtin_amdgcn_wmma_f32_16x16x32_bf16(false, a0, false, bv2, (short)0, acc[0][2], false, false);
        acc[1][2] = __builtin_amdgcn_wmma_f32_16x16x32_bf16(false, a1, false, bv2, (short)0, acc[1][2], false, false);
        acc[0][3] = __builtin_amdgcn_wmma_f32_16x16x32_bf16(false, a0, false, bv3, (short)0, acc[0][3], false, false);
        acc[1][3] = __builtin_amdgcn_wmma_f32_16x16x32_bf16(false, a1, false, bv3, (short)0, acc[1][3], false, false);
    }

#pragma unroll
    for (int t = 0; t < 4; ++t) {
        int col = colbase + t * 16 + m;
        float bias = b2[c * D_H + col];
        float s = 0.f;
#pragma unroll
        for (int a = 0; a < 2; ++a) {
#pragma unroll
            for (int v = 0; v < 8; ++v) {
                int row = tm * 32 + a * 16 + v + 8 * hh;
                if (row < n_c) s += relu_f(acc[a][t][v] + bias);
            }
        }
        s += __shfl_down(s, 16, 32);              // combine the two half-wave row groups
        if (hh == 0)
            part[((size_t)c * TILES_M32 + tm) * D_H + col] = s;
    }
}

// ---------------- tail: pooled means + attention head + classifier ----------------
__global__ void k_tail(const float* __restrict__ part, const int* __restrict__ cnt,
                       const float* __restrict__ Wfc, const float* __restrict__ bfc,
                       const float* __restrict__ Wa, const float* __restrict__ ba,
                       const float* __restrict__ Wb, const float* __restrict__ bb,
                       const float* __restrict__ Wc, const float* __restrict__ bc,
                       const float* __restrict__ Wr, const float* __restrict__ br,
                       const float* __restrict__ Wcls, const float* __restrict__ bcls,
                       float* __restrict__ out) {
    __shared__ float pooled[C_CLUST][D_H];
    __shared__ float hs[C_CLUST][D_H];
    __shared__ float gates[C_CLUST][D_ATT];
    __shared__ float attw[C_CLUST];
    __shared__ float hp[D_H];
    __shared__ float hr[D_ATT];
    int tid = threadIdx.x;   // blockDim = 512

    for (int c = 0; c < C_CLUST; ++c) {
        int nc = cnt[c];
        int tiles = (nc + 31) / 32;
        float s = 0.f;
        for (int t = 0; t < tiles; ++t)
            s += part[((size_t)c * TILES_M32 + t) * D_H + tid];
        pooled[c][tid] = s / (float)(nc > 0 ? nc : 1);
    }
    __syncthreads();
    for (int c = 0; c < C_CLUST; ++c) {
        float a = bfc[tid];
        for (int k = 0; k < D_H; ++k) a += pooled[c][k] * Wfc[k * D_H + tid];
        hs[c][tid] = relu_f(a);
    }
    __syncthreads();
    if (tid < D_ATT) {
        for (int c = 0; c < C_CLUST; ++c) {
            float aa = ba[tid], ab = bb[tid];
            for (int k = 0; k < D_H; ++k) {
                float hv = hs[c][k];
                aa += hv * Wa[k * D_ATT + tid];
                ab += hv * Wb[k * D_ATT + tid];
            }
            gates[c][tid] = tanhf(aa) * (1.f / (1.f + expf(-ab)));
        }
    }
    __syncthreads();
    if (tid == 0) {
        float la[C_CLUST];
        float mx = -1e30f;
        for (int c = 0; c < C_CLUST; ++c) {
            float s = bc[0];
            for (int j = 0; j < D_ATT; ++j) s += gates[c][j] * Wc[j];
            la[c] = s;
            if (s > mx) mx = s;
        }
        float den = 0.f;
        for (int c = 0; c < C_CLUST; ++c) { attw[c] = expf(la[c] - mx); den += attw[c]; }
        for (int c = 0; c < C_CLUST; ++c) attw[c] /= den;
    }
    __syncthreads();
    {
        float s = 0.f;
        for (int c = 0; c < C_CLUST; ++c) s += attw[c] * hs[c][tid];
        hp[tid] = s;
    }
    __syncthreads();
    if (tid < D_ATT) {
        float a = br[tid];
        for (int k = 0; k < D_H; ++k) a += hp[k] * Wr[k * D_ATT + tid];
        hr[tid] = relu_f(a);
    }
    __syncthreads();
    if (tid == 0) {
        float lg[N_CLS];
        float mx = -1e30f;
        int arg = 0;
        for (int j = 0; j < N_CLS; ++j) {
            float s = bcls[j];
            for (int k = 0; k < D_ATT; ++k) s += hr[k] * Wcls[k * N_CLS + j];
            lg[j] = s;
            out[j] = s;
            if (s > mx) { mx = s; arg = j; }
        }
        float den = 0.f, e[N_CLS];
        for (int j = 0; j < N_CLS; ++j) { e[j] = expf(lg[j] - mx); den += e[j]; }
        for (int j = 0; j < N_CLS; ++j) out[4 + j] = e[j] / den;
        out[8] = (float)arg;
    }
}

extern "C" void kernel_launch(void* const* d_in, const int* in_sizes, int n_in,
                              void* d_out, int out_size, void* d_ws, size_t ws_size,
                              hipStream_t stream) {
    const float* x    = (const float*)d_in[0];
    const int*   cid  = (const int*)d_in[1];
    const float* W1   = (const float*)d_in[2];
    const float* b1   = (const float*)d_in[3];
    const float* W2   = (const float*)d_in[4];
    const float* b2   = (const float*)d_in[5];
    const float* Wfc  = (const float*)d_in[6];
    const float* bfc  = (const float*)d_in[7];
    const float* Wa   = (const float*)d_in[8];
    const float* ba   = (const float*)d_in[9];
    const float* Wb   = (const float*)d_in[10];
    const float* bb   = (const float*)d_in[11];
    const float* Wc   = (const float*)d_in[12];
    const float* bc   = (const float*)d_in[13];
    const float* Wr   = (const float*)d_in[14];
    const float* br   = (const float*)d_in[15];
    const float* Wcls = (const float*)d_in[16];
    const float* bcls = (const float*)d_in[17];
    float* out = (float*)d_out;

    char* ws = (char*)d_ws;
    bf16_t* xb   = (bf16_t*)(ws + OFF_XB);
    bf16_t* w1t  = (bf16_t*)(ws + OFF_W1T);
    bf16_t* w2t  = (bf16_t*)(ws + OFF_W2T);
    int*    idxl = (int*)(ws + OFF_IDX);
    int*    cnt  = (int*)(ws + OFF_CNT);
    int*    basep= (int*)(ws + OFF_BASE);
    bf16_t* h1   = (bf16_t*)(ws + OFF_H1);
    float*  part = (float*)(ws + OFF_PART);

    {
        int n = N_PTS * D_IN;
        k_cvt_x<<<(n + 255) / 256, 256, 0, stream>>>(x, xb, n);
    }
    {
        int n = C_CLUST * D_H * D_IN;
        k_cvt_w1<<<(n + 255) / 256, 256, 0, stream>>>(W1, w1t);
    }
    {
        int n = C_CLUST * D_H * D_H;
        k_cvt_w2<<<(n + 255) / 256, 256, 0, stream>>>(W2, w2t);
    }
    k_bucket<<<C_CLUST, 256, 0, stream>>>(cid, idxl, cnt);
    k_base<<<1, 32, 0, stream>>>(cnt, basep);
    k_phi1<<<dim3(2, TILES_M32, C_CLUST), 128, 0, stream>>>(xb, w1t, b1, idxl, cnt, basep, h1);
    k_phi2<<<dim3(2, TILES_M32, C_CLUST), 128, 0, stream>>>(h1, w2t, b2, cnt, basep, part);
    k_tail<<<1, 512, 0, stream>>>(part, cnt, Wfc, bfc, Wa, ba, Wb, bb, Wc, bc,
                                  Wr, br, Wcls, bcls, out);
}
